// Model_9088150798642
// MI455X (gfx1250) — matrix operand order repaired and checked by reading, not compiled, not run on hardware
//
#include <hip/hip_runtime.h>
#include <hip/hip_bf16.h>
#include <math.h>

typedef __attribute__((ext_vector_type(16))) _Float16 v16h;
typedef __attribute__((ext_vector_type(8)))  _Float16 v8h;
typedef __attribute__((ext_vector_type(8)))  float    v8f;

#define TSTEPS 512
#define HID    64
#define CIN    5
#define CCONV  32

// ---------------------------------------------------------------------------
// Kernel 1: conv1d center-tap + PReLU for batch rows 0..15, all timesteps.
// Output layout: pre[((t*16)+b)*32 + o] as f16  (A-fragment friendly).
// ---------------------------------------------------------------------------
__global__ void conv_prep_kernel(const float* __restrict__ x,
                                 const float* __restrict__ conv_w,
                                 const float* __restrict__ conv_b,
                                 const float* __restrict__ prelu1_a,
                                 _Float16* __restrict__ pre) {
    int idx = blockIdx.x * blockDim.x + threadIdx.x;     // 512*16*32 = 262144
    if (idx >= TSTEPS * 16 * CCONV) return;
    int o = idx & 31;
    int b = (idx >> 5) & 15;
    int t = idx >> 9;
    const float* xr = x + ((size_t)b * TSTEPS + t) * CIN;   // x[b][t][:]
    float s = conv_b[o];
#pragma unroll
    for (int k = 0; k < CIN; ++k)
        s += xr[k] * conv_w[o * 15 + k * 3 + 1];            // center tap
    float a = prelu1_a[0];
    s = (s >= 0.f) ? s : a * s;
    pre[idx] = (_Float16)s;
}

// ---------------------------------------------------------------------------
// Branch-free activations: v_exp_f32 + v_rcp_f32, no exec divergence,
// correct saturation at +/-inf.
// ---------------------------------------------------------------------------
__device__ __forceinline__ float fast_rcp(float x) {
    return __builtin_amdgcn_rcpf(x);
}
__device__ __forceinline__ float sigf(float x) {
    return fast_rcp(1.f + __expf(-x));
}
__device__ __forceinline__ float tanh_fast(float x) {
    return 1.f - 2.f * fast_rcp(1.f + __expf(2.f * x));
}

// ---------------------------------------------------------------------------
// Kernel 2: 512-step LSTM recurrence (layer 0, batches 0..15) via WMMA,
// then PReLU + linear(2x64) + tanh epilogue for batch 0.
// One block, 4 waves. Wave w owns hidden columns [16w, 16w+16).
// Ping-pong h buffers in LDS -> single barrier per timestep.
// ---------------------------------------------------------------------------
__global__ __launch_bounds__(128)
void lstm_rec_kernel(const _Float16* __restrict__ pre,
                     const float* __restrict__ w_ih,   // (256,32)
                     const float* __restrict__ w_hh,   // (256,64)
                     const float* __restrict__ b_ih,
                     const float* __restrict__ b_hh,
                     const float* __restrict__ prelu2_a,
                     const float* __restrict__ lin_w,  // (2,64)
                     const float* __restrict__ lin_b,  // (2,)
                     float* __restrict__ out) {
    __shared__ _Float16 lds_h[2][16 * HID];   // ping-pong h[batch][hid] f16
    __shared__ float    h0buf[HID];           // final h of batch 0 (f32)

    const int tid  = threadIdx.x;
    const int wave = tid >> 5;
    const int lane = tid & 31;
    const int m    = lane & 15;            // A row (batch) / B column-in-tile
    const int hi   = lane >> 4;            // lane half
    const int kbA  = hi ? 8 : 0;           // A-frag K base (16x32 f16 layout)
    const int kbB  = hi ? 16 : 0;          // unused; B-frag uses kbA table order

    // ---- preload weight B-fragments + biases (constant over time) ----
    v16h  bwih[4];
    v16h  bwhh[4][2];
    float biasv[4];
#pragma unroll
    for (int g = 0; g < 4; ++g) {
        const int col = g * HID + wave * 16 + m;    // gate column 0..255
        biasv[g] = b_ih[col] + b_hh[col];
        const float* wr = w_ih + (size_t)col * 32;
        v16h t{};
#pragma unroll
        for (int e = 0; e < 16; ++e) t[e] = (_Float16)wr[kbA + (e < 8 ? e : e + 8)];
        bwih[g] = t;
        const float* hr = w_hh + (size_t)col * HID;
#pragma unroll
        for (int c = 0; c < 2; ++c) {
            v16h u{};
#pragma unroll
            for (int e = 0; e < 16; ++e) u[e] = (_Float16)hr[c * 32 + kbA + (e < 8 ? e : e + 8)];
            bwhh[g][c] = u;
        }
    }

    // ---- init state ----
    v8f cfrag = {0.f, 0.f, 0.f, 0.f, 0.f, 0.f, 0.f, 0.f};
    float hv0 = 0.f;                        // last h for batch0, this wave's col
    for (int i = tid; i < 16 * HID; i += 128) lds_h[0][i] = (_Float16)0.f;
    __syncthreads();

    // ---- recurrence: read lds_h[t&1], write lds_h[(t+1)&1], 1 barrier ----
    const int wcol = wave * 16 + m;
    for (int t = 0; t < TSTEPS; ++t) {
        const int rb = t & 1;
        const int wb = rb ^ 1;

        // A-fragment: x_t  (16x32 f16 layout: K = kbA + {e | e+8})
        const _Float16* pr = pre + ((size_t)t * 16 + m) * 32 + kbA;
        v8h xlo = *(const v8h*)(pr);
        v8h xhi = *(const v8h*)(pr + 16);
        v16h a_x = __builtin_shufflevector(xlo, xhi,
            0, 1, 2, 3, 4, 5, 6, 7, 8, 9, 10, 11, 12, 13, 14, 15);

        // A-fragments: h_{t-1} (16x64 -> two 16x32 chunks) from LDS
        const _Float16* hrow = &lds_h[rb][m * HID];
        v8h h00 = *(const v8h*)(hrow + kbA);
        v8h h01 = *(const v8h*)(hrow + kbA + 16);
        v16h a_h0 = __builtin_shufflevector(h00, h01,
            0, 1, 2, 3, 4, 5, 6, 7, 8, 9, 10, 11, 12, 13, 14, 15);
        v8h h10 = *(const v8h*)(hrow + 32 + kbA);
        v8h h11 = *(const v8h*)(hrow + 32 + kbA + 16);
        v16h a_h1 = __builtin_shufflevector(h10, h11,
            0, 1, 2, 3, 4, 5, 6, 7, 8, 9, 10, 11, 12, 13, 14, 15);

        // gates: bias -> + Xt*WihT -> + H*WhhT (f32 accumulate)
        v8f gacc[4];
#pragma unroll
        for (int g = 0; g < 4; ++g) {
            float bv = biasv[g];
            v8f acc = {bv, bv, bv, bv, bv, bv, bv, bv};
            acc = __builtin_amdgcn_wmma_f32_16x16x32_f16(
                false, a_x,  false, bwih[g],    (short)0, acc, false, false);
            acc = __builtin_amdgcn_wmma_f32_16x16x32_f16(
                false, a_h0, false, bwhh[g][0], (short)0, acc, false, false);
            acc = __builtin_amdgcn_wmma_f32_16x16x32_f16(
                false, a_h1, false, bwhh[g][1], (short)0, acc, false, false);
            gacc[g] = acc;
        }

        // LSTM cell update (elementwise on matching fragment slots)
#pragma unroll
        for (int j = 0; j < 8; ++j) {
            float iv = sigf(gacc[0][j]);
            float fv = sigf(gacc[1][j]);
            float gv = tanh_fast(gacc[2][j]);
            float ov = sigf(gacc[3][j]);
            float cn = fv * cfrag[j] + iv * gv;
            cfrag[j] = cn;
            float hn = ov * tanh_fast(cn);
            int row = hi ? (j + 8) : j;                 // batch row of slot j
            lds_h[wb][row * HID + wcol] = (_Float16)hn;
            if (!hi && j == 0) hv0 = hn;                // batch 0 value
        }
        __syncthreads();   // h_t visible before next iteration's reads
    }

    // ---- epilogue: PReLU + linear(2x64) + tanh on batch 0 ----
    if (lane < 16) h0buf[wave * 16 + lane] = hv0;
    __syncthreads();
    if (tid == 0) {
        float p2 = prelu2_a[0];
#pragma unroll
        for (int o = 0; o < 2; ++o) {
            float s = lin_b[o];
            for (int j = 0; j < HID; ++j) {
                float a = h0buf[j];
                a = (a >= 0.f) ? a : p2 * a;
                s += lin_w[o * HID + j] * a;
            }
            out[o] = tanh_fast(s);
        }
    }
}

// ---------------------------------------------------------------------------
extern "C" void kernel_launch(void* const* d_in, const int* in_sizes, int n_in,
                              void* d_out, int out_size, void* d_ws, size_t ws_size,
                              hipStream_t stream) {
    const float* x        = (const float*)d_in[0];
    const float* conv_w   = (const float*)d_in[1];
    const float* conv_b   = (const float*)d_in[2];
    const float* prelu1_a = (const float*)d_in[3];
    const float* w_ih0    = (const float*)d_in[4];
    const float* w_hh0    = (const float*)d_in[5];
    const float* b_ih0    = (const float*)d_in[6];
    const float* b_hh0    = (const float*)d_in[7];
    const float* prelu2_a = (const float*)d_in[16];
    const float* lin_w    = (const float*)d_in[17];
    const float* lin_b    = (const float*)d_in[18];
    float* out = (float*)d_out;

    _Float16* pre = (_Float16*)d_ws;   // 512*16*32 f16 = 512 KB

    const int nelem = TSTEPS * 16 * CCONV;
    conv_prep_kernel<<<(nelem + 255) / 256, 256, 0, stream>>>(
        x, conv_w, conv_b, prelu1_a, pre);

    lstm_rec_kernel<<<1, 128, 0, stream>>>(
        pre, w_ih0, w_hh0, b_ih0, b_hh0, prelu2_a, lin_w, lin_b, out);
}
